// protmamba_66821101191412
// MI455X (gfx1250) — matrix operand order repaired and checked by reading, not compiled, not run on hardware
//
#include <hip/hip_runtime.h>
#include <hip/hip_bf16.h>

// ---------------- model dims ----------------
#define BATCH   4
#define SEQLEN  1024
#define VOCAB   28
#define D_MODEL 1024
#define N_LAYERS 4
#define D_INNER 2048
#define D_STATE 16
#define DT_RANK 64
#define D_CONV  4
#define NUM_CLS 2
#define NTOK    (BATCH * SEQLEN)      // 4096

typedef __bf16 bf16;
typedef bf16  v8bf  __attribute__((ext_vector_type(8)));
typedef bf16  v16bf __attribute__((ext_vector_type(16)));
typedef float v8f   __attribute__((ext_vector_type(8)));

// ---- gfx1250 async global->LDS path (guarded; fallback = sync LDS stage) ----
#if __has_builtin(__builtin_amdgcn_global_load_async_to_lds_b128) && \
    __has_builtin(__builtin_amdgcn_s_wait_asynccnt)
#define PM_ASYNC 1
typedef int pm_v4i __attribute__((ext_vector_type(4)));
typedef __attribute__((address_space(1))) pm_v4i pm_gv4i;   // global int4
typedef __attribute__((address_space(3))) pm_v4i pm_lv4i;   // LDS int4
// generic->LDS is a low-32-bit truncation on amdgcn; generic->global is identity
#define PM_AS1(p) ((pm_gv4i*)(unsigned long long)(p))
#define PM_AS3(p) ((pm_lv4i*)(unsigned int)(unsigned long long)(p))
#endif

// ===================================================================
// embedding + positional encoding
// ===================================================================
__global__ void pm_embed_k(const int* __restrict__ x, const float* __restrict__ emb,
                           const float* __restrict__ pos, float* __restrict__ h) {
    int i = blockIdx.x * blockDim.x + threadIdx.x;          // over NTOK*D_MODEL
    if (i >= NTOK * D_MODEL) return;
    int d = i % D_MODEL;
    int t = i / D_MODEL;                                     // b*SEQLEN + l
    int l = t % SEQLEN;
    int tok = x[t];
    h[i] = emb[tok * D_MODEL + d] + pos[l * D_MODEL + d];
}

// ===================================================================
// rmsnorm (per token) -> bf16
// ===================================================================
__global__ void pm_rmsnorm_k(const float* __restrict__ h, const float* __restrict__ w,
                             bf16* __restrict__ out) {
    __shared__ float red[256];
    int t = blockIdx.x;
    int tid = threadIdx.x;
    const float* hp = h + (size_t)t * D_MODEL;
    float s = 0.f;
    for (int i = tid; i < D_MODEL; i += 256) { float v = hp[i]; s += v * v; }
    red[tid] = s;
    __syncthreads();
    for (int off = 128; off > 0; off >>= 1) {
        if (tid < off) red[tid] += red[tid + off];
        __syncthreads();
    }
    float inv = rsqrtf(red[0] / (float)D_MODEL + 1e-5f);
    for (int i = tid; i < D_MODEL; i += 256)
        out[(size_t)t * D_MODEL + i] = (bf16)(hp[i] * inv * w[i]);
}

// ===================================================================
// f32 -> bf16 cast
// ===================================================================
__global__ void pm_cast_k(const float* __restrict__ src, bf16* __restrict__ dst, int n) {
    int i = blockIdx.x * blockDim.x + threadIdx.x;
    if (i < n) dst[i] = (bf16)src[i];
}

// ===================================================================
// WMMA GEMM:  C[M,N] = act(A[M,K] * W[N,K]^T + bias)
//   block = 8 waves = 128(M) x 64(N) tile; each wave a 16x64 slice.
//   B tile (64 cols x 32 K, bf16) double-buffered in LDS, staged with
//   GLOBAL_LOAD_ASYNC_TO_LDS_B128 (ASYNCcnt) and shared by all 8 waves.
//   A fragments: per-wave direct global loads (distinct rows, pipelined).
//   act: 0 none, 1 relu, 2 softplus ;  accum: C += result (no bias/act)
// ===================================================================
__global__ __launch_bounds__(256)
void pm_gemm_wmma_k(const bf16* __restrict__ A, int lda,
                    const bf16* __restrict__ W, int ldw,
                    const float* __restrict__ bias,
                    float* __restrict__ C, int ldc,
                    int M, int N, int K, int act, int accum) {
    __shared__ bf16 sB[2][64][32];                 // 2 x 4KB

    const int tid  = threadIdx.x;
    const int wave = tid >> 5;
    const int lane = tid & 31;
    const int hl   = lane >> 4;                    // wave half: 0 or 1
    const int l15  = lane & 15;
    const int n0   = blockIdx.x * 64;
    const int m0   = blockIdx.y * 128 + wave * 16;

    // ---- B stage mapping: thread tid copies one 16B chunk per K-step ----
    const int ccol = tid >> 2;                     // 0..63 (block-local col)
    const int cchk = tid & 3;                      // 0..3  (16B chunks)
    int gcol = n0 + ccol; if (gcol >= N) gcol = N - 1;
    const bf16* gsrc = W + (size_t)gcol * ldw + cchk * 8;

#ifdef PM_ASYNC
#define PM_STAGE(buf, kk)                                                     \
    __builtin_amdgcn_global_load_async_to_lds_b128(                           \
        PM_AS1(gsrc + (kk)), PM_AS3(&sB[(buf)][ccol][cchk * 8]), 0, 0)
#else
#define PM_STAGE(buf, kk)                                                     \
    (*(v8bf*)&sB[(buf)][ccol][cchk * 8] = *(const v8bf*)(gsrc + (kk)))
#endif

    v8f acc0 = {}, acc1 = {}, acc2 = {}, acc3 = {};
    const bf16* arowp = A + (size_t)(m0 + l15) * lda + hl * 16;

    PM_STAGE(0, 0);

    const int nk = K / 32;
    for (int kt = 0; kt < nk; ++kt) {
        const int kk  = kt * 32;
        const int cur = kt & 1;

        if (kt + 1 < nk) {
            PM_STAGE(cur ^ 1, kk + 32);
#ifdef PM_ASYNC
            __builtin_amdgcn_s_wait_asynccnt(1);   // cur buffer's copy retired
#endif
        } else {
#ifdef PM_ASYNC
            __builtin_amdgcn_s_wait_asynccnt(0);
#endif
        }
        __syncthreads();                           // all waves' slices visible

        union { v16bf v; v8bf h[2]; } af;
        af.h[0] = *(const v8bf*)(arowp + kk);
        af.h[1] = *(const v8bf*)(arowp + kk + 8);
        __builtin_prefetch(arowp + kk + 32, 0, 1);

        v16bf b0 = *(const v16bf*)&sB[cur][0 * 16 + l15][hl * 16];
        v16bf b1 = *(const v16bf*)&sB[cur][1 * 16 + l15][hl * 16];
        v16bf b2 = *(const v16bf*)&sB[cur][2 * 16 + l15][hl * 16];
        v16bf b3 = *(const v16bf*)&sB[cur][3 * 16 + l15][hl * 16];

        acc0 = __builtin_amdgcn_wmma_f32_16x16x32_bf16(false, af.v, false, b0,
                                                       (short)0, acc0, false, false);
        acc1 = __builtin_amdgcn_wmma_f32_16x16x32_bf16(false, af.v, false, b1,
                                                       (short)0, acc1, false, false);
        acc2 = __builtin_amdgcn_wmma_f32_16x16x32_bf16(false, af.v, false, b2,
                                                       (short)0, acc2, false, false);
        acc3 = __builtin_amdgcn_wmma_f32_16x16x32_bf16(false, af.v, false, b3,
                                                       (short)0, acc3, false, false);

        __syncthreads();                           // before buffer reuse
    }
#undef PM_STAGE

    // epilogue: VGPR r holds row m0 + hl*8 + r, col n0 + nt*16 + l15
    #pragma unroll
    for (int nt = 0; nt < 4; ++nt) {
        const v8f* accp = (nt == 0) ? &acc0 : (nt == 1) ? &acc1 : (nt == 2) ? &acc2 : &acc3;
        int col = n0 + nt * 16 + l15;
        if (col >= N) continue;
        #pragma unroll
        for (int r = 0; r < 8; ++r) {
            int row = m0 + hl * 8 + r;
            size_t off = (size_t)row * ldc + col;
            float v = (*accp)[r];
            if (accum) {
                C[off] += v;
            } else {
                if (bias) v += bias[col];
                if (act == 1) v = v > 0.f ? v : 0.f;
                else if (act == 2) v = v > 20.f ? v : log1pf(__expf(v));  // softplus
                C[off] = v;
            }
        }
    }
}

// ===================================================================
// depthwise causal conv (D_CONV=4) + bias + SiLU ; writes fp32 + bf16
// ===================================================================
__global__ void pm_conv_silu_k(const float* __restrict__ xz,
                               const float* __restrict__ cw, const float* __restrict__ cb,
                               float* __restrict__ xcf, bf16* __restrict__ xcb) {
    int i = blockIdx.x * blockDim.x + threadIdx.x;   // over NTOK*D_INNER
    if (i >= NTOK * D_INNER) return;
    int d = i % D_INNER;
    int t = i / D_INNER;
    int l = t % SEQLEN;
    int b = t / SEQLEN;
    float acc = cb[d];
    #pragma unroll
    for (int j = 0; j < D_CONV; ++j) {
        int ll = l - (D_CONV - 1) + j;
        if (ll >= 0)
            acc += cw[d * D_CONV + j] * xz[((size_t)(b * SEQLEN + ll)) * (2 * D_INNER) + d];
    }
    float s = acc / (1.f + __expf(-acc));            // silu
    xcf[i] = s;
    xcb[i] = (bf16)s;
}

// ===================================================================
// extract delta low-rank columns (0..DT_RANK-1 of dbc) -> bf16
// ===================================================================
__global__ void pm_extract_dt_k(const float* __restrict__ dbc, bf16* __restrict__ out) {
    int i = blockIdx.x * blockDim.x + threadIdx.x;   // over NTOK*DT_RANK
    if (i >= NTOK * DT_RANK) return;
    int r = i % DT_RANK;
    int t = i / DT_RANK;
    out[i] = (bf16)dbc[(size_t)t * (DT_RANK + 2 * D_STATE) + r];
}

// ===================================================================
// selective scan + D skip + SiLU(z) gating -> bf16 y_gated
// ===================================================================
__global__ __launch_bounds__(256)
void pm_scan_gate_k(const float* __restrict__ delta, const float* __restrict__ xc,
                    const float* __restrict__ dbc, const float* __restrict__ xz,
                    const float* __restrict__ A_log, const float* __restrict__ Dp,
                    bf16* __restrict__ yg) {
    __shared__ float sB[D_STATE];
    __shared__ float sC[D_STATE];
    const int d = blockIdx.x * 256 + threadIdx.x;    // channel
    const int b = blockIdx.y;
    const int tid = threadIdx.x;

    float Arow[D_STATE];
    #pragma unroll
    for (int n = 0; n < D_STATE; ++n) Arow[n] = -__expf(A_log[d * D_STATE + n]);
    const float dval = Dp[d];
    float h[D_STATE];
    #pragma unroll
    for (int n = 0; n < D_STATE; ++n) h[n] = 0.f;

    for (int l = 0; l < SEQLEN; ++l) {
        size_t t = (size_t)b * SEQLEN + l;
        if (tid < 2 * D_STATE) {
            float v = dbc[t * (DT_RANK + 2 * D_STATE) + DT_RANK + tid];
            if (tid < D_STATE) sB[tid] = v; else sC[tid - D_STATE] = v;
        }
        __syncthreads();

        float de  = delta[t * D_INNER + d];
        float xcv = xc[t * D_INNER + d];
        float dx  = de * xcv;
        float y = 0.f;
        #pragma unroll
        for (int n = 0; n < D_STATE; ++n) {
            h[n] = __expf(de * Arow[n]) * h[n] + dx * sB[n];
            y += h[n] * sC[n];
        }
        y += dval * xcv;
        float z = xz[t * (2 * D_INNER) + D_INNER + d];
        float zg = z / (1.f + __expf(-z));           // silu(z)
        yg[t * D_INNER + d] = (bf16)(y * zg);
        __syncthreads();
    }
}

// ===================================================================
// scalar GEMM (tiny N / tiny M cases): C = act(A*W^T + bias)
// ===================================================================
__global__ void pm_gemm_scalar_k(const float* __restrict__ A, int lda,
                                 const float* __restrict__ W,
                                 const float* __restrict__ bias,
                                 float* __restrict__ C, int ldc,
                                 int M, int N, int K, int act) {
    int i = blockIdx.x * blockDim.x + threadIdx.x;
    if (i >= M * N) return;
    int n = i % N, m = i / N;
    const float* ap = A + (size_t)m * lda;
    const float* wp = W + (size_t)n * K;
    float s = bias ? bias[n] : 0.f;
    for (int k = 0; k < K; ++k) s += ap[k] * wp[k];
    if (act == 1) s = s > 0.f ? s : 0.f;
    C[(size_t)m * ldc + n] = s;
}

// ===================================================================
// launch
// ===================================================================
extern "C" void kernel_launch(void* const* d_in, const int* in_sizes, int n_in,
                              void* d_out, int out_size, void* d_ws, size_t ws_size,
                              hipStream_t stream) {
    const int*   x         = (const int*)  d_in[0];
    const float* embedding = (const float*)d_in[1];
    const float* pos_enc   = (const float*)d_in[2];
    const float* norm_w    = (const float*)d_in[3];
    const float* in_proj_w = (const float*)d_in[4];
    const float* conv_w    = (const float*)d_in[5];
    const float* conv_b    = (const float*)d_in[6];
    const float* x_proj_w  = (const float*)d_in[7];
    const float* dt_proj_w = (const float*)d_in[8];
    const float* dt_proj_b = (const float*)d_in[9];
    const float* A_log     = (const float*)d_in[10];
    const float* Dp        = (const float*)d_in[11];
    const float* out_proj_w= (const float*)d_in[12];
    const float* cls_fc1_w = (const float*)d_in[13];
    const float* cls_fc1_b = (const float*)d_in[14];
    const float* cls_fc2_w = (const float*)d_in[15];
    const float* cls_fc2_b = (const float*)d_in[16];
    const float* dec_fc1_w = (const float*)d_in[17];
    const float* dec_fc1_b = (const float*)d_in[18];
    const float* dec_fc2_w = (const float*)d_in[19];
    const float* dec_fc2_b = (const float*)d_in[20];
    float* out = (float*)d_out;

    // -------- workspace carve-up --------
    char* ws = (char*)d_ws;
    size_t off = 0;
    auto carve = [&](size_t bytes) { char* p = ws + off; off += (bytes + 255) & ~(size_t)255; return p; };
    float* W0_h     = (float*)carve((size_t)NTOK * D_MODEL * 4);        // residual stream
    bf16*  W1a_xn   = (bf16*) carve((size_t)NTOK * D_MODEL * 2);        // normed / h bf16
    bf16*  W1b_act  = (bf16*) carve((size_t)NTOK * D_INNER * 2);        // xc / y_gated bf16
    float* W2_xz    = (float*)carve((size_t)NTOK * 2 * D_INNER * 4);    // in_proj out
    float* W3_xcf   = (float*)carve((size_t)NTOK * D_INNER * 4);        // conv+silu fp32 / dec hidden
    float* W4_dbc   = (float*)carve((size_t)NTOK * (DT_RANK + 2 * D_STATE) * 4);
    bf16*  W4b_dtlr = (bf16*) carve((size_t)NTOK * DT_RANK * 2);
    float* W5_delta = (float*)carve((size_t)NTOK * D_INNER * 4);
    bf16*  Wbf      = (bf16*) carve((size_t)(2 * D_INNER) * D_MODEL * 2); // weight conversion
    float* Wcls     = (float*)carve((size_t)BATCH * D_MODEL * 4);
    (void)ws_size; (void)n_in; (void)in_sizes; (void)out_size;

    auto cdiv = [](int a, int b) { return (a + b - 1) / b; };

    // -------- embedding + pos --------
    pm_embed_k<<<cdiv(NTOK * D_MODEL, 256), 256, 0, stream>>>(x, embedding, pos_enc, W0_h);

    // -------- layers --------
    for (int l = 0; l < N_LAYERS; ++l) {
        const float* nw  = norm_w    + (size_t)l * D_MODEL;
        const float* ipw = in_proj_w + (size_t)l * 2 * D_INNER * D_MODEL;
        const float* cw  = conv_w    + (size_t)l * D_INNER * D_CONV;
        const float* cb  = conv_b    + (size_t)l * D_INNER;
        const float* xpw = x_proj_w  + (size_t)l * (DT_RANK + 2 * D_STATE) * D_INNER;
        const float* dtw = dt_proj_w + (size_t)l * D_INNER * DT_RANK;
        const float* dtb = dt_proj_b + (size_t)l * D_INNER;
        const float* al  = A_log     + (size_t)l * D_INNER * D_STATE;
        const float* dp  = Dp        + (size_t)l * D_INNER;
        const float* opw = out_proj_w+ (size_t)l * D_MODEL * D_INNER;

        // rmsnorm -> bf16
        pm_rmsnorm_k<<<NTOK, 256, 0, stream>>>(W0_h, nw, W1a_xn);

        // in_proj: xz = xn * ipw^T   (M=4096, N=4096, K=1024)
        pm_cast_k<<<cdiv(2 * D_INNER * D_MODEL, 256), 256, 0, stream>>>(ipw, Wbf, 2 * D_INNER * D_MODEL);
        pm_gemm_wmma_k<<<dim3(cdiv(2 * D_INNER, 64), NTOK / 128), 256, 0, stream>>>(
            W1a_xn, D_MODEL, Wbf, D_MODEL, nullptr, W2_xz, 2 * D_INNER,
            NTOK, 2 * D_INNER, D_MODEL, 0, 0);

        // conv + silu
        pm_conv_silu_k<<<cdiv(NTOK * D_INNER, 256), 256, 0, stream>>>(W2_xz, cw, cb, W3_xcf, W1b_act);

        // x_proj: dbc = xc * xpw^T  (M=4096, N=96, K=2048)
        pm_cast_k<<<cdiv((DT_RANK + 2 * D_STATE) * D_INNER, 256), 256, 0, stream>>>(
            xpw, Wbf, (DT_RANK + 2 * D_STATE) * D_INNER);
        pm_gemm_wmma_k<<<dim3(cdiv(DT_RANK + 2 * D_STATE, 64), NTOK / 128), 256, 0, stream>>>(
            W1b_act, D_INNER, Wbf, D_INNER, nullptr, W4_dbc, DT_RANK + 2 * D_STATE,
            NTOK, DT_RANK + 2 * D_STATE, D_INNER, 0, 0);

        // dt_proj + softplus: delta = softplus(dlr * dtw^T + dtb)  (M=4096, N=2048, K=64)
        pm_extract_dt_k<<<cdiv(NTOK * DT_RANK, 256), 256, 0, stream>>>(W4_dbc, W4b_dtlr);
        pm_cast_k<<<cdiv(D_INNER * DT_RANK, 256), 256, 0, stream>>>(dtw, Wbf, D_INNER * DT_RANK);
        pm_gemm_wmma_k<<<dim3(cdiv(D_INNER, 64), NTOK / 128), 256, 0, stream>>>(
            W4b_dtlr, DT_RANK, Wbf, DT_RANK, dtb, W5_delta, D_INNER,
            NTOK, D_INNER, DT_RANK, 2, 0);

        // selective scan + gate -> bf16 y_gated (overwrites W1b_act)
        pm_scan_gate_k<<<dim3(D_INNER / 256, BATCH), 256, 0, stream>>>(
            W5_delta, W3_xcf, W4_dbc, W2_xz, al, dp, W1b_act);

        // out_proj, accumulate into residual: h += yg * opw^T (M=4096, N=1024, K=2048)
        pm_cast_k<<<cdiv(D_MODEL * D_INNER, 256), 256, 0, stream>>>(opw, Wbf, D_MODEL * D_INNER);
        pm_gemm_wmma_k<<<dim3(cdiv(D_MODEL, 64), NTOK / 128), 256, 0, stream>>>(
            W1b_act, D_INNER, Wbf, D_INNER, nullptr, W0_h, D_MODEL,
            NTOK, D_MODEL, D_INNER, 0, 1);
    }

    // -------- decoder MLP: relu(h*fc1^T+b1)*fc2^T+b2 --------
    pm_cast_k<<<cdiv(NTOK * D_MODEL, 256), 256, 0, stream>>>(W0_h, W1a_xn, NTOK * D_MODEL);
    pm_cast_k<<<cdiv(D_MODEL * D_MODEL, 256), 256, 0, stream>>>(dec_fc1_w, Wbf, D_MODEL * D_MODEL);
    pm_gemm_wmma_k<<<dim3(cdiv(D_MODEL, 64), NTOK / 128), 256, 0, stream>>>(
        W1a_xn, D_MODEL, Wbf, D_MODEL, dec_fc1_b, W3_xcf, D_MODEL,
        NTOK, D_MODEL, D_MODEL, 1, 0);
    pm_gemm_scalar_k<<<cdiv(NTOK * VOCAB, 256), 256, 0, stream>>>(
        W3_xcf, D_MODEL, dec_fc2_w, dec_fc2_b, out, VOCAB,
        NTOK, VOCAB, D_MODEL, 0);

    // -------- classifier MLP on last token --------
    pm_gemm_scalar_k<<<cdiv(BATCH * D_MODEL, 256), 256, 0, stream>>>(
        W0_h + (size_t)(SEQLEN - 1) * D_MODEL, SEQLEN * D_MODEL,
        cls_fc1_w, cls_fc1_b, Wcls, D_MODEL, BATCH, D_MODEL, D_MODEL, 1);
    pm_gemm_scalar_k<<<1, 256, 0, stream>>>(
        Wcls, D_MODEL, cls_fc2_w, cls_fc2_b, out + (size_t)NTOK * VOCAB, NUM_CLS,
        BATCH, NUM_CLS, D_MODEL, 0);
}
